// USPAttention_62654982914823
// MI455X (gfx1250) — matrix operand order repaired and checked by reading, not compiled, not run on hardware
//
#include <hip/hip_runtime.h>

typedef __attribute__((ext_vector_type(16))) _Float16 v16h;
typedef __attribute__((ext_vector_type(8)))  _Float16 v8h;
typedef __attribute__((ext_vector_type(2)))  __fp16   h2;   // return type of cvt_pkrtz
typedef __attribute__((ext_vector_type(8)))  float    v8f;

constexpr int SEQ   = 4096;
constexpr int HDIM  = 1024;   // heads * d
constexpr int NH    = 16;
constexpr int D     = 64;     // head dim
constexpr int BM    = 128;    // q rows per workgroup (16 per wave x 8 waves)
constexpr int BN    = 64;     // k/v rows per tile
constexpr int NT    = SEQ / BN;
constexpr int PITCH = 80;     // f16 pitch: 160 bytes/row, 32B aligned, bank-friendly

// scale * log2(e): softmax computed in base-2 domain; folded into Q conversion
#define TSCALE (0.125f * 1.44269504088896340736f)

__global__ __launch_bounds__(256)
void fa_fwd_kernel(const float* __restrict__ Q, const float* __restrict__ K,
                   const float* __restrict__ V, float* __restrict__ O)
{
    __shared__ __align__(32) _Float16 ldsK [BN * PITCH];          // row-major [krow][dim]
    __shared__ __align__(32) _Float16 ldsVt[BN * PITCH + 64];     // transposed [dim][vrow], +rot pad
    __shared__ __align__(32) _Float16 ldsP [8 * 16 * PITCH];      // per-wave prob scratch

    const int tid  = threadIdx.x;
    const int wave = tid >> 5;
    const int lane = tid & 31;
    const int l16  = lane & 15;
    const int half = lane >> 4;          // 0: lanes 0-15, 1: lanes 16-31

    const int head = blockIdx.y;
    const int q0   = blockIdx.x * BM + wave * 16;   // this wave's 16 q rows

    // ---- Load Q once into A-fragment layout, pre-scaled by TSCALE ----
    v16h aq[2];
    {
        const float* qrow = Q + (size_t)(q0 + l16) * HDIM + head * D;
        #pragma unroll
        for (int f = 0; f < 2; ++f) {
            const int d0 = f * 32;
            #pragma unroll
            for (int e = 0; e < 8; ++e) {
                aq[f][e]     = (_Float16)(qrow[d0 + 16 * half + e] * TSCALE);
                aq[f][8 + e] = (_Float16)(qrow[d0 + 16 * half + 8 + e] * TSCALE);
            }
        }
    }

    // all-ones B fragment: accumulates row sums of P via WMMA (no shuffle reduce)
    v16h bones;
    #pragma unroll
    for (int e = 0; e < 16; ++e) bones[e] = (_Float16)1.0f;

    // ---- accumulators / softmax state (rows m = r + 8*half) ----
    v8f o_acc[4];
    #pragma unroll
    for (int c = 0; c < 4; ++c) o_acc[c] = (v8f){};
    v8f o_sum = (v8f){};                 // row sums (every lane of a half holds a copy)
    float mrow[8];
    #pragma unroll
    for (int r = 0; r < 8; ++r) mrow[r] = -3.0e38f;

    // cooperative fill assignment: 4 threads per row, 16 dims each
    const int frow = tid >> 2;           // 0..63 : k/v row
    const int fcol = (tid & 3) * 16;     // 0,16,32,48 : dim chunk

    _Float16* Pw = &ldsP[wave * 16 * PITCH];

    // ---- software pipeline: prefetch K/V tile into registers ----
    float4 kreg[4], vreg[4];
    {
        const size_t gb = (size_t)frow * HDIM + head * D + fcol;
        #pragma unroll
        for (int i = 0; i < 4; ++i) {
            kreg[i] = *(const float4*)(K + gb + i * 4);
            vreg[i] = *(const float4*)(V + gb + i * 4);
        }
    }

    for (int kt = 0; kt < NT; ++kt) {
        // ---- convert prefetched registers -> LDS (packed f16) ----
        #pragma unroll
        for (int i = 0; i < 4; ++i) {
            h2 k01 = __builtin_amdgcn_cvt_pkrtz(kreg[i].x, kreg[i].y);
            h2 k23 = __builtin_amdgcn_cvt_pkrtz(kreg[i].z, kreg[i].w);
            _Float16* kd = &ldsK[frow * PITCH + fcol + i * 4];
            *(h2*)(kd)     = k01;
            *(h2*)(kd + 2) = k23;
            h2 v01 = __builtin_amdgcn_cvt_pkrtz(vreg[i].x, vreg[i].y);
            h2 v23 = __builtin_amdgcn_cvt_pkrtz(vreg[i].z, vreg[i].w);
            // transposed V: Vt index = dim*PITCH + (dim&48 rotation) + vrow
            _Float16* vt = &ldsVt[(fcol + i * 4) * PITCH + fcol + frow];
            vt[0 * PITCH] = (_Float16)v01[0];
            vt[1 * PITCH] = (_Float16)v01[1];
            vt[2 * PITCH] = (_Float16)v23[0];
            vt[3 * PITCH] = (_Float16)v23[1];
        }
        __syncthreads();

        // ---- issue next tile's global loads unconditionally (clamped index:
        //      last iteration harmlessly re-reads the final tile; keeps the
        //      loop body branch-free so the compiler doesn't peel it) ----
        {
            const int ktn = (kt + 1 < NT) ? (kt + 1) : (NT - 1);
            const size_t gb = (size_t)(ktn * BN + frow) * HDIM + head * D + fcol;
            #pragma unroll
            for (int i = 0; i < 4; ++i) {
                kreg[i] = *(const float4*)(K + gb + i * 4);
                vreg[i] = *(const float4*)(V + gb + i * 4);
            }
        }

        // ---- S = Q * K^T : 4 column chunks of 16, reduction over D in 2x32 ----
        v8f s[4];
        #pragma unroll
        for (int c = 0; c < 4; ++c) s[c] = (v8f){};
        #pragma unroll
        for (int f = 0; f < 2; ++f) {
            const int d0 = f * 32;
            #pragma unroll
            for (int c = 0; c < 4; ++c) {
                const v16h bk = *(const v16h*)&ldsK[(c * 16 + l16) * PITCH + d0 + 16 * half];
                s[c] = __builtin_amdgcn_wmma_f32_16x16x32_f16(
                           false, aq[f], false, bk, (short)0, s[c], false, false);
            }
        }

        // ---- online softmax (base-2 domain; scale folded into Q) ----
        #pragma unroll
        for (int r = 0; r < 8; ++r) {
            float mx = fmaxf(fmaxf(s[0][r], s[1][r]), fmaxf(s[2][r], s[3][r]));
            #pragma unroll
            for (int m = 8; m >= 1; m >>= 1)
                mx = fmaxf(mx, __shfl_xor(mx, m, 32));   // stays within 16-lane half
            const float mnew = fmaxf(mrow[r], mx);
            const float corr = __builtin_amdgcn_exp2f(mrow[r] - mnew);
            mrow[r] = mnew;
            o_sum[r] *= corr;
            #pragma unroll
            for (int c = 0; c < 4; c += 2) {
                const float p0 = __builtin_amdgcn_exp2f(s[c][r] - mnew);
                const float p1 = __builtin_amdgcn_exp2f(s[c + 1][r] - mnew);
                o_acc[c][r]     *= corr;
                o_acc[c + 1][r] *= corr;
                const h2 pp = __builtin_amdgcn_cvt_pkrtz(p0, p1);
                _Float16* pd = &Pw[(r + 8 * half) * PITCH + l16];
                pd[c * 16]       = (_Float16)pp[0];
                pd[(c + 1) * 16] = (_Float16)pp[1];
            }
        }

        // ---- O += P * V  (plus ones-column for row sums) ----
        #pragma unroll
        for (int f = 0; f < 2; ++f) {
            const int j0 = f * 32;
            const v8h lo = *(const v8h*)&Pw[l16 * PITCH + j0 + 16 * half];
            const v8h hi = *(const v8h*)&Pw[l16 * PITCH + j0 + 16 * half + 8];
            v16h ap;
            #pragma unroll
            for (int e = 0; e < 8; ++e) { ap[e] = lo[e]; ap[8 + e] = hi[e]; }
            #pragma unroll
            for (int c = 0; c < 4; ++c) {
                const v16h bv = *(const v16h*)
                    &ldsVt[(c * 16 + l16) * PITCH + c * 16 + j0 + 16 * half];
                o_acc[c] = __builtin_amdgcn_wmma_f32_16x16x32_f16(
                               false, ap, false, bv, (short)0, o_acc[c], false, false);
            }
            // row-sum accumulation: P x ones
            o_sum = __builtin_amdgcn_wmma_f32_16x16x32_f16(
                        false, ap, false, bones, (short)0, o_sum, false, false);
        }
        __syncthreads();   // LDS reads complete before next tile's fill
    }

    // ---- epilogue: O / rowsum, store fp32 (row m=r+8*half, col=l16+16c) ----
    float* obase = O + (size_t)q0 * HDIM + head * D;
    #pragma unroll
    for (int r = 0; r < 8; ++r) {
        const float inv_l = 1.0f / o_sum[r];
        const int m = r + 8 * half;
        #pragma unroll
        for (int c = 0; c < 4; ++c)
            obase[(size_t)m * HDIM + c * 16 + l16] = o_acc[c][r] * inv_l;
    }
}

extern "C" void kernel_launch(void* const* d_in, const int* in_sizes, int n_in,
                              void* d_out, int out_size, void* d_ws, size_t ws_size,
                              hipStream_t stream) {
    const float* q = (const float*)d_in[0];
    const float* k = (const float*)d_in[1];
    const float* v = (const float*)d_in[2];
    float* out = (float*)d_out;
    dim3 grid(SEQ / BM, NH);
    fa_fwd_kernel<<<grid, dim3(256), 0, stream>>>(q, k, v, out);
}
